// MixHop_82231443849291
// MI455X (gfx1250) — matrix operand order repaired and checked by reading, not compiled, not run on hardware
//
#include <hip/hip_runtime.h>
#include <hip/hip_bf16.h>

typedef __bf16 bf16_t;
typedef __attribute__((ext_vector_type(16))) __bf16 v16bf;
typedef __attribute__((ext_vector_type(8))) float v8f;

#define F_IN 128
#define HID 192
#define OUT_F 64
#define HID_LD 200   // padded LDS row stride (bf16 elements) to break bank alignment

// ---------------- degree / normalization ----------------

__global__ void k_set_deg(float* deg, int n) {
    int i = blockIdx.x * blockDim.x + threadIdx.x;
    if (i < n) deg[i] = 1.0f;   // self loop contributes 1
}

__global__ void k_deg_edges(const long long* __restrict__ ei, float* deg, int E) {
    int e = blockIdx.x * blockDim.x + threadIdx.x;
    if (e < E) {
        int col = (int)ei[(size_t)E + e];   // target node
        unsafeAtomicAdd(&deg[col], 1.0f);
    }
}

__global__ void k_dinv(float* deg, int n) {
    int i = blockIdx.x * blockDim.x + threadIdx.x;
    if (i < n) {
        float d = deg[i];
        deg[i] = d > 0.0f ? rsqrtf(d) : 0.0f;   // in-place deg -> dinv
    }
}

// ---------------- propagation (gather-scale-scatter) ----------------

// self-loop term: out[i,:] = dinv[i]^2 * h[i,:]
__global__ void k_self_prop(const float* __restrict__ h, const float* __restrict__ dinv,
                            float* __restrict__ out, int total) {
    int i = blockIdx.x * blockDim.x + threadIdx.x;
    if (i < total) {
        int node = i >> 7;          // /128
        float di = dinv[node];
        out[i] = di * di * h[i];
    }
}

// edge term: out[col,:] += dinv[row]*dinv[col] * h[row,:]
// 32 threads per edge, 4 floats per thread; all operands are L2-resident on MI455X.
__global__ void k_edge_prop(const long long* __restrict__ ei, const float* __restrict__ dinv,
                            const float* __restrict__ h, float* __restrict__ out, int E) {
    long long gid = (long long)blockIdx.x * blockDim.x + threadIdx.x;
    if (gid >= (long long)E * 32) return;
    int e = (int)(gid >> 5);
    int c = ((int)gid & 31) * 4;
    int row = (int)ei[e];
    int col = (int)ei[(size_t)E + e];
    float nrm = dinv[row] * dinv[col];
    const float4 v = *(const float4*)(h + (size_t)row * F_IN + c);
    float* dst = out + (size_t)col * F_IN + c;
    unsafeAtomicAdd(dst + 0, nrm * v.x);
    unsafeAtomicAdd(dst + 1, nrm * v.y);
    unsafeAtomicAdd(dst + 2, nrm * v.z);
    unsafeAtomicAdd(dst + 3, nrm * v.w);
}

// ---------------- weight packing: fp32 W[K,64] -> bf16 B-fragment order ----------------
// Contiguous-k fragment order used here, matching the A loaders (16-bit B, 32x16):
//   lane l (0-15): col n=l,   elements e=0..15 -> K = k0 + e
//   lane l (16-31): col n=l-16, elements e=0..15 -> K = k0 + 16 + e
// Packed index: (((k0/32)*64 + n)*2 + khalf)*16 + e
__global__ void k_pack_w(const float* __restrict__ W, bf16_t* __restrict__ Wp, int K) {
    int idx = blockIdx.x * blockDim.x + threadIdx.x;
    if (idx >= K * 64) return;
    int e      = idx & 15;
    int khalf  = (idx >> 4) & 1;
    int n      = (idx >> 5) & 63;
    int kblock = idx >> 11;
    int k = kblock * 32 + khalf * 16 + e;
    Wp[idx] = (bf16_t)W[k * 64 + n];
}

// ---------------- WMMA fragment loaders ----------------

// A fragment (16x32 bf16) from row-major fp32 [16,ld] tile, convert on the fly.
// Lane l: row M = l&15; kb = (l>=16)?16:0; elements 0..15 -> K=kb..kb+15
__device__ inline v16bf load_a_frag_g(const float* __restrict__ p, int ld, int lane) {
    int r  = lane & 15;
    int kb = (lane >> 4) << 4;
    const float* p0 = p + r * ld + kb;
    v16bf a;
#pragma unroll
    for (int i = 0; i < 8; ++i) {
        a[i]     = (bf16_t)p0[i];
        a[i + 8] = (bf16_t)p0[8 + i];
    }
    return a;
}

// A fragment from bf16 LDS hidden tile (row stride HID_LD)
__device__ inline v16bf load_a_frag_lds(const bf16_t* p, int lane) {
    int r  = lane & 15;
    int kb = (lane >> 4) << 4;
    const bf16_t* p0 = p + r * HID_LD + kb;
    v16bf a;
#pragma unroll
    for (int i = 0; i < 8; ++i) {
        a[i]     = p0[i];
        a[i + 8] = p0[8 + i];
    }
    return a;
}

// B fragment from packed weights: one contiguous 32B load per lane
__device__ inline v16bf load_b_frag(const bf16_t* __restrict__ Wp, int k0, int n0, int lane) {
    int idx = ((((k0 >> 5) << 6) + n0 + (lane & 15)) * 2 + (lane >> 4)) << 4;
    return *(const v16bf*)(Wp + idx);
}

// ---------------- fused MLP: 3 hop projections + ReLU + output GEMM ----------------

__global__ void __launch_bounds__(256)
k_mixhop_mlp(const float* __restrict__ x, const float* __restrict__ h1,
             const float* __restrict__ h2,
             const bf16_t* __restrict__ Wp0, const bf16_t* __restrict__ Wp1,
             const bf16_t* __restrict__ Wp2, const bf16_t* __restrict__ WpO,
             const float* __restrict__ b0, const float* __restrict__ b1,
             const float* __restrict__ b2, const float* __restrict__ bout,
             float* __restrict__ out, int ntiles) {
    __shared__ bf16_t hid_s[8][16 * HID_LD];
    int lane = threadIdx.x & 31;
    int wave = threadIdx.x >> 5;
    int tile = blockIdx.x * 8 + wave;
    if (tile >= ntiles) return;          // whole-wave exit: EXEC stays all-ones for WMMA
    int row0 = tile * 16;
    bf16_t* hid = &hid_s[wave][0];

    const float*  srcs[3] = {x, h1, h2};
    const bf16_t* wps[3]  = {Wp0, Wp1, Wp2};
    const float*  bs[3]   = {b0, b1, b2};

    int nsel  = lane & 15;               // D-matrix column N per lane
    int mBase = (lane >> 4) << 3;        // D-matrix row base (VGPR r -> M = mBase + r)

    // Stage 1: per-hop [16,128]@[128,64] + bias -> ReLU -> LDS [16,192] as bf16
#pragma unroll
    for (int hop = 0; hop < 3; ++hop) {
        const float* src = srcs[hop] + (size_t)row0 * F_IN;
        v8f acc[4];
#pragma unroll
        for (int t = 0; t < 4; ++t) {
            float bv = bs[hop][t * 16 + nsel];
#pragma unroll
            for (int r = 0; r < 8; ++r) acc[t][r] = bv;
        }
#pragma unroll
        for (int k0 = 0; k0 < F_IN; k0 += 32) {
            v16bf a = load_a_frag_g(src + k0, F_IN, lane);
#pragma unroll
            for (int t = 0; t < 4; ++t) {
                v16bf b = load_b_frag(wps[hop], k0, t * 16, lane);
                acc[t] = __builtin_amdgcn_wmma_f32_16x16x32_bf16(
                             false, a, false, b, (short)0, acc[t], false, false);
            }
        }
#pragma unroll
        for (int t = 0; t < 4; ++t) {
#pragma unroll
            for (int r = 0; r < 8; ++r) {
                float v = acc[t][r];
                v = v > 0.0f ? v : 0.0f;                           // ReLU
                hid[(mBase + r) * HID_LD + hop * 64 + t * 16 + nsel] = (bf16_t)v;
            }
        }
    }

    // Stage 2: [16,192]@[192,64] + bout -> global
    v8f acc2[4];
#pragma unroll
    for (int t = 0; t < 4; ++t) {
        float bv = bout[t * 16 + nsel];
#pragma unroll
        for (int r = 0; r < 8; ++r) acc2[t][r] = bv;
    }
#pragma unroll
    for (int k0 = 0; k0 < HID; k0 += 32) {
        v16bf a = load_a_frag_lds(hid + k0, lane);
#pragma unroll
        for (int t = 0; t < 4; ++t) {
            v16bf b = load_b_frag(WpO, k0, t * 16, lane);
            acc2[t] = __builtin_amdgcn_wmma_f32_16x16x32_bf16(
                          false, a, false, b, (short)0, acc2[t], false, false);
        }
    }
#pragma unroll
    for (int t = 0; t < 4; ++t) {
#pragma unroll
        for (int r = 0; r < 8; ++r) {
            out[(size_t)(row0 + mBase + r) * OUT_F + t * 16 + nsel] = acc2[t][r];
        }
    }
}

// ---------------- host side ----------------

extern "C" void kernel_launch(void* const* d_in, const int* in_sizes, int n_in,
                              void* d_out, int out_size, void* d_ws, size_t ws_size,
                              hipStream_t stream) {
    const float*     x    = (const float*)d_in[0];
    const long long* ei   = (const long long*)d_in[1];   // int64 [2,E]
    const float*     W0   = (const float*)d_in[2];
    const float*     b0   = (const float*)d_in[3];
    const float*     W1   = (const float*)d_in[4];
    const float*     b1   = (const float*)d_in[5];
    const float*     W2   = (const float*)d_in[6];
    const float*     b2   = (const float*)d_in[7];
    const float*     Wout = (const float*)d_in[8];
    const float*     bout = (const float*)d_in[9];
    float* out = (float*)d_out;

    int N = in_sizes[0] / F_IN;   // 50000
    int E = in_sizes[1] / 2;      // 800000

    // workspace layout (256B aligned slices)
    char* ws = (char*)d_ws;
    size_t off = 0;
    auto alloc = [&](size_t bytes) {
        size_t o = off;
        off += (bytes + 255) & ~(size_t)255;
        return o;
    };
    float*  deg = (float*)(ws + alloc((size_t)N * 4));            // -> becomes dinv in place
    float*  h1  = (float*)(ws + alloc((size_t)N * F_IN * 4));
    float*  h2  = (float*)(ws + alloc((size_t)N * F_IN * 4));
    bf16_t* Wp0 = (bf16_t*)(ws + alloc(128 * 64 * 2));
    bf16_t* Wp1 = (bf16_t*)(ws + alloc(128 * 64 * 2));
    bf16_t* Wp2 = (bf16_t*)(ws + alloc(128 * 64 * 2));
    bf16_t* WpO = (bf16_t*)(ws + alloc(192 * 64 * 2));

    const int B = 256;

    // pack weights to bf16 fragment order (independent of props)
    k_pack_w<<<(128 * 64 + B - 1) / B, B, 0, stream>>>(W0, Wp0, 128);
    k_pack_w<<<(128 * 64 + B - 1) / B, B, 0, stream>>>(W1, Wp1, 128);
    k_pack_w<<<(128 * 64 + B - 1) / B, B, 0, stream>>>(W2, Wp2, 128);
    k_pack_w<<<(192 * 64 + B - 1) / B, B, 0, stream>>>(Wout, WpO, 192);

    // degrees -> dinv
    k_set_deg  <<<(N + B - 1) / B, B, 0, stream>>>(deg, N);
    k_deg_edges<<<(E + B - 1) / B, B, 0, stream>>>(ei, deg, E);
    k_dinv     <<<(N + B - 1) / B, B, 0, stream>>>(deg, N);

    int totF = N * F_IN;
    long long eThreads = (long long)E * 32;
    int eBlocks = (int)((eThreads + B - 1) / B);

    // h1 = prop(x)
    k_self_prop<<<(totF + B - 1) / B, B, 0, stream>>>(x, deg, h1, totF);
    k_edge_prop<<<eBlocks, B, 0, stream>>>(ei, deg, x, h1, E);
    // h2 = prop(h1)
    k_self_prop<<<(totF + B - 1) / B, B, 0, stream>>>(h1, deg, h2, totF);
    k_edge_prop<<<eBlocks, B, 0, stream>>>(ei, deg, h1, h2, E);

    // fused MLP (WMMA)
    int ntiles = (N + 15) / 16;           // 3125, exact
    int grid = (ntiles + 7) / 8;
    k_mixhop_mlp<<<grid, 256, 0, stream>>>(x, h1, h2, Wp0, Wp1, Wp2, WpO,
                                           b0, b1, b2, bout, out, ntiles);
}
